// TrueSlidingWindowAttention_61795989455014
// MI455X (gfx1250) — compile-verified, matrix operand order checked by reading, not run on hardware
//
#include <hip/hip_runtime.h>
#include <hip/hip_bf16.h>

typedef __attribute__((ext_vector_type(16))) _Float16 v16h;
typedef __attribute__((ext_vector_type(8)))  float    v8f;

#define SEQ        4096
#define DHEAD      64
#define WINDOW     512
#define ROWS_BLK   64     // query rows per workgroup (4 waves x 16)
#define ROWS_WAVE  16
#define KTILE      32     // keys per inner tile == WMMA K for f16

// pitches in halves; dword pitch p with p%8==4 -> rows 16B-aligned (b128
// merging) AND p*n mod 64 distinct over 16 lanes (conflict-free row reads)
#define KT_PITCH   72     // 36 dwords
#define VT_PITCH   40     // 20 dwords
#define PT_PITCH   40     // 20 dwords

// A-fragment element -> contraction index (ISA 7.12.2, 16-bit A 16x32):
// lane half h (lane>>4), element e in 0..15
__device__ __forceinline__ int kloc(int e, int h) {
    return (e & 7) + ((e >> 3) << 4) + (h << 3);
}

__launch_bounds__(128)
__global__ void swa_fwd_kernel(const float* __restrict__ q,
                               const float* __restrict__ k,
                               const float* __restrict__ v,
                               float* __restrict__ out) {
    // double-buffered K tile [key][d] and transposed V tile [d][key]
    __shared__ __align__(16) _Float16 kt[2][KTILE][KT_PITCH];
    __shared__ __align__(16) _Float16 vt_t[2][DHEAD][VT_PITCH];
    // per-wave P staging (C-layout -> A-layout)
    __shared__ __align__(16) _Float16 pt[4][ROWS_WAVE][PT_PITCH];

    const int tid   = threadIdx.x;
    const int wv    = tid >> 5;
    const int lane  = tid & 31;
    const int h     = lane >> 4;
    const int nl    = lane & 15;

    const int batch = blockIdx.y;
    const int r0    = blockIdx.x * ROWS_BLK;
    const int r0w   = r0 + wv * ROWS_WAVE;

    const float scale = 0.125f; // d^-0.5 = 1/8

    const float* qb = q + (size_t)batch * SEQ * DHEAD;
    const float* kb = k + (size_t)batch * SEQ * DHEAD;
    const float* vb = v + (size_t)batch * SEQ * DHEAD;

    // Cooperative-stage mapping: thread owns key-row srow, 16 d-cols from scol
    const int srow = tid >> 2;          // 0..31
    const int scol = (tid & 3) * 16;    // 0,16,32,48

    auto stage = [&](int kb0, int buf) {
        const float4* kp = (const float4*)&kb[(size_t)(kb0 + srow) * DHEAD + scol];
        const float4* vp = (const float4*)&vb[(size_t)(kb0 + srow) * DHEAD + scol];
        float kvals[16], vvals[16];
        #pragma unroll
        for (int i = 0; i < 4; ++i) {
            *(float4*)&kvals[i * 4] = kp[i];
            *(float4*)&vvals[i * 4] = vp[i];
        }
        #pragma unroll
        for (int i = 0; i < 16; ++i)          // 2x ds_store_b128
            kt[buf][srow][scol + i] = (_Float16)kvals[i];
        #pragma unroll
        for (int i = 0; i < 16; ++i)          // transposed scatter
            vt_t[buf][scol + i][srow] = (_Float16)vvals[i];
    };

    // ---- Q A-fragments (two K=32 chunks over d=64), pre-scaled ----
    v16h qa0, qa1;
    #pragma unroll
    for (int e = 0; e < 16; ++e) {
        const int kl = kloc(e, h);
        qa0[e] = (_Float16)(qb[(size_t)(r0w + nl) * DHEAD + kl]      * scale);
        qa1[e] = (_Float16)(qb[(size_t)(r0w + nl) * DHEAD + 32 + kl] * scale);
    }
    // constant ones B-fragment: P x 1 -> row sums in C-layout
    v16h vone;
    #pragma unroll
    for (int e = 0; e < 16; ++e) vone[e] = (_Float16)1.0f;

    // ---- flash-attention state (C-layout rows: m = r + 8*h) ----
    v8f o[4];
    #pragma unroll
    for (int dc = 0; dc < 4; ++dc) o[dc] = (v8f){};
    float mrow[8], lrow[8];
    #pragma unroll
    for (int r = 0; r < 8; ++r) { mrow[r] = -1.0e30f; lrow[r] = 0.0f; }

    // Uniform key range for the whole block (so __syncthreads is legal)
    int kb_lo = r0 - (WINDOW - 1);
    if (kb_lo < 0) kb_lo = 0;
    kb_lo &= ~(KTILE - 1);
    const int kb_hi = r0 + ROWS_BLK - 1;
    const int ntiles = (kb_hi - kb_lo) / KTILE + 1;

    stage(kb_lo, 0);
    __syncthreads();

    for (int it = 0; it < ntiles; ++it) {
        const int kb0 = kb_lo + it * KTILE;
        const int buf = it & 1;

        // stage next tile into the other buffer (loads fly under compute)
        if (it + 1 < ntiles) stage(kb0 + KTILE, buf ^ 1);

        // Wave-uniform tile classification (scalar branches, EXEC untouched).
        const bool all_invalid = (kb0 > r0w + ROWS_WAVE - 1) ||
                                 (kb0 + KTILE - 1 < r0w - (WINDOW - 1));
        if (!all_invalid) {
            const bool all_valid = (kb0 + KTILE - 1 <= r0w) &&
                                   (kb0 >= r0w + ROWS_WAVE - 1 - (WINDOW - 1));

            // ---- S = Q * K^T : 16x32 tile as two 16x16 halves ----
            v8f s[2];
            #pragma unroll
            for (int jh = 0; jh < 2; ++jh) {
                v16h b0, b1;
                #pragma unroll
                for (int e = 0; e < 16; ++e) {
                    const int kl = kloc(e, h);
                    b0[e] = kt[buf][jh * 16 + nl][kl];        // B[d][j] = K[j][d]
                    b1[e] = kt[buf][jh * 16 + nl][32 + kl];
                }
                v8f acc = (v8f){};
                acc = __builtin_amdgcn_wmma_f32_16x16x32_f16(false, qa0, false, b0,
                                                             (short)0, acc, false, false);
                acc = __builtin_amdgcn_wmma_f32_16x16x32_f16(false, qa1, false, b1,
                                                             (short)0, acc, false, false);
                s[jh] = acc;
            }

            // ---- sliding-window mask (boundary tiles only) + online softmax ----
            float p0[8], p1[8], tred[8];
            if (all_valid) {
                #pragma unroll
                for (int r = 0; r < 8; ++r) {
                    p0[r] = s[0][r];
                    p1[r] = s[1][r];
                    tred[r] = fmaxf(p0[r], p1[r]);
                }
            } else {
                #pragma unroll
                for (int r = 0; r < 8; ++r) {
                    const int rowg = r0w + r + 8 * h;
                    const int c0   = kb0 + nl;
                    const int c1   = kb0 + 16 + nl;
                    const bool ok0 = (c0 <= rowg) && (c0 >= rowg - (WINDOW - 1));
                    const bool ok1 = (c1 <= rowg) && (c1 >= rowg - (WINDOW - 1));
                    p0[r] = ok0 ? s[0][r] : -3.0e38f;
                    p1[r] = ok1 ? s[1][r] : -3.0e38f;
                    tred[r] = fmaxf(p0[r], p1[r]);
                }
            }
            // row-max across the 16 lanes of each half (max has no matmul trick)
            #pragma unroll
            for (int off = 8; off >= 1; off >>= 1)
                #pragma unroll
                for (int r = 0; r < 8; ++r)
                    tred[r] = fmaxf(tred[r], __shfl_xor(tred[r], off, 32));

            float alpha[8];
            #pragma unroll
            for (int r = 0; r < 8; ++r) {
                const float mnew = fmaxf(mrow[r], tred[r]);
                alpha[r] = __expf(mrow[r] - mnew);
                mrow[r]  = mnew;
                p0[r]    = __expf(p0[r] - mnew);
                p1[r]    = __expf(p1[r] - mnew);
            }

            // ---- P: C-layout -> A-layout via wave-private LDS ----
            #pragma unroll
            for (int r = 0; r < 8; ++r) {
                pt[wv][r + 8 * h][nl]      = (_Float16)p0[r];
                pt[wv][r + 8 * h][16 + nl] = (_Float16)p1[r];
            }
            asm volatile("s_wait_dscnt 0" ::: "memory");   // CDNA5 split DS counter
            __builtin_amdgcn_wave_barrier();

            v16h pa;
            #pragma unroll
            for (int e = 0; e < 16; ++e)
                pa[e] = pt[wv][nl][kloc(e, h)];

            // ---- row sums via WMMA: csum = P * ones (C[m][n] = sum_k P[m][k]) ----
            v8f csum = (v8f){};
            csum = __builtin_amdgcn_wmma_f32_16x16x32_f16(false, pa, false, vone,
                                                          (short)0, csum, false, false);
            #pragma unroll
            for (int r = 0; r < 8; ++r)
                lrow[r] = lrow[r] * alpha[r] + csum[r];

            // ---- O = alpha*O + P * V : 4 d-chunks, K=32 contraction ----
            #pragma unroll
            for (int dc = 0; dc < 4; ++dc) {
                v16h bvf;
                #pragma unroll
                for (int e = 0; e < 16; ++e)
                    bvf[e] = vt_t[buf][dc * 16 + nl][kloc(e, h)]; // B[key][n] = V^T[n][key]
                v8f oc = o[dc];
                #pragma unroll
                for (int r = 0; r < 8; ++r) oc[r] *= alpha[r];
                o[dc] = __builtin_amdgcn_wmma_f32_16x16x32_f16(false, pa, false, bvf,
                                                               (short)0, oc, false, false);
            }
        }
        __syncthreads();   // separates this iter's reads from iter+2's writes
    }

    // ---- epilogue: O * (1/l), C-layout scatter ----
    float linv[8];
    #pragma unroll
    for (int r = 0; r < 8; ++r) linv[r] = 1.0f / lrow[r];
    #pragma unroll
    for (int dc = 0; dc < 4; ++dc) {
        #pragma unroll
        for (int r = 0; r < 8; ++r) {
            const int rowg = r0w + r + 8 * h;
            out[((size_t)batch * SEQ + rowg) * DHEAD + dc * 16 + nl] =
                o[dc][r] * linv[r];
        }
    }
}

extern "C" void kernel_launch(void* const* d_in, const int* in_sizes, int n_in,
                              void* d_out, int out_size, void* d_ws, size_t ws_size,
                              hipStream_t stream) {
    const float* q = (const float*)d_in[0];
    const float* k = (const float*)d_in[1];
    const float* v = (const float*)d_in[2];
    float* out = (float*)d_out;

    const int b = in_sizes[0] / (SEQ * DHEAD);   // 16
    dim3 grid(SEQ / ROWS_BLK, b);
    dim3 block(128);
    hipLaunchKernelGGL(swa_fwd_kernel, grid, block, 0, stream, q, k, v, out);
}
